// LSTM_89489938579751
// MI455X (gfx1250) — matrix operand order repaired and checked by reading, not compiled, not run on hardware
//
#include <hip/hip_runtime.h>
#include <hip/hip_bf16.h>
#include <stdint.h>

#define B_   128
#define S_   512
#define V_   128
#define H_   1024
#define E_   256
#define C_   10
#define N4_  4096
#define NBLK 64        // persistent workgroups (grid-sync'd); each owns 16 hidden cols
#define HCOLS 16
#define LROWS 64       // 4 gates x 16 hidden cols staged in LDS
#define LSTRIDE 1032   // padded row stride (halves): 2064B -> lane l hits banks 4l..4l+3

typedef __attribute__((ext_vector_type(16))) __bf16 v16bf;
typedef __attribute__((ext_vector_type(8)))  float  v8f;

static __device__ __forceinline__ uint16_t f32_to_bf16(float f) {
  uint32_t u = __float_as_uint(f);
  u += 0x7FFFu + ((u >> 16) & 1u);   // round-to-nearest-even
  return (uint16_t)(u >> 16);
}

// A fragment (16-bit, 16x32): lane holds 16 contiguous halves at k+base,
// base = (lane<16 ? 0 : 16). Two 16B loads.
static __device__ __forceinline__ v16bf load_a16(const uint16_t* p) {
  union { uint4 q[2]; v16bf v; } t;
  t.q[0] = *(const uint4*)(p);
  t.q[1] = *(const uint4*)(p + 8);
  return t.v;
}
// B fragment (32x16): lane holds 16 contiguous K halves of its column (32B).
static __device__ __forceinline__ v16bf load_b32(const uint16_t* p) {
  union { uint4 q[2]; v16bf v; } t;
  t.q[0] = *(const uint4*)(p);
  t.q[1] = *(const uint4*)(p + 8);
  return t.v;
}

#define WMMA_BF16(A, Bm, Cm) \
  __builtin_amdgcn_wmma_f32_16x16x32_bf16(false, (A), false, (Bm), (short)0, (Cm), false, false)

// ---------------- prep kernels ----------------
__global__ void k_prep_wx(const float* __restrict__ g0, const float* __restrict__ g1,
                          const float* __restrict__ g2, const float* __restrict__ g3,
                          uint16_t* __restrict__ WxT) {
  int i = blockIdx.x * 256 + threadIdx.x;   // i = n*E_ + e,  n in [0,4096)
  int n = i >> 8;
  int e = i & 255;
  int gate = n >> 10;
  int col  = n & 1023;
  const float* W = gate == 0 ? g0 : gate == 1 ? g1 : gate == 2 ? g2 : g3;
  WxT[i] = f32_to_bf16(W[e * H_ + col]);    // (E,H) row-major -> (N,K)
}

__global__ void k_prep_wh(const float* __restrict__ g0, const float* __restrict__ g1,
                          const float* __restrict__ g2, const float* __restrict__ g3,
                          uint16_t* __restrict__ WhT) {
  int i = blockIdx.x * 256 + threadIdx.x;   // i = n*H_ + k
  int n = i >> 10;
  int k = i & 1023;
  int gate = n >> 10;
  int col  = n & 1023;
  const float* W = gate == 0 ? g0 : gate == 1 ? g1 : gate == 2 ? g2 : g3;
  WhT[i] = f32_to_bf16(W[k * H_ + col]);
}

__global__ void k_prep_misc(const float* __restrict__ emb, uint16_t* __restrict__ embb,
                            uint16_t* __restrict__ hbf, float* __restrict__ hf32,
                            float* __restrict__ cst, unsigned* __restrict__ cnt) {
  int i = blockIdx.x * 256 + threadIdx.x;   // 131072 = B_*H_ threads
  if (i < V_ * E_) embb[i] = f32_to_bf16(emb[i]);
  if (i < B_ * H_) {
    hbf[i] = 0; hbf[i + B_ * H_] = 0;       // both h double-buffers
    hf32[i] = 0.0f;
    cst[i]  = 0.0f;
  }
  if (i == 0) *cnt = 0u;
}

// ---------------- persistent LSTM kernel ----------------
__global__ __launch_bounds__(256, 1)
void k_lstm(const int* __restrict__ xTok,
            const uint16_t* __restrict__ embb,
            const uint16_t* __restrict__ WxT,
            const uint16_t* __restrict__ WhT,
            const float* __restrict__ bg, const float* __restrict__ bi,
            const float* __restrict__ bf_, const float* __restrict__ bo,
            uint16_t* __restrict__ hbf,      // 2 buffers of B_*H_ bf16
            float* __restrict__ hf32,
            float* __restrict__ cst,
            unsigned* __restrict__ cnt) {
  __shared__ uint16_t sWh[LROWS * LSTRIDE];   // ~129 KB of the 320 KB WGP LDS

  const int tid    = threadIdx.x;
  const int wave   = tid >> 5;
  const int lane   = tid & 31;
  const int laneHi = lane >> 4;
  const int lane16 = lane & 15;
  const int m0     = wave * 16;                 // wave's 16 batch rows
  const int mrow   = m0 + lane16;               // A row this lane fetches
  const int hc0    = blockIdx.x * HCOLS;
  const int hc     = hc0 + lane16;              // this lane's hidden column

  // ---- stage this block's Wh tile into LDS (once; reused for 512 steps) ----
  {
    const int j   = tid >> 2;                   // LDS row 0..63 (gate*16 + col)
    const int sub = tid & 3;                    // 4 threads per row
    const int gate = j >> 4, c = j & 15;
    const uint16_t* src = WhT + (size_t)(gate * H_ + hc0 + c) * H_;
#pragma unroll
    for (int q = sub * 32; q < sub * 32 + 32; ++q)   // 32 x 16B per thread
      *(uint4*)(&sWh[(size_t)j * LSTRIDE + q * 8]) = *(const uint4*)(src + q * 8);
  }
  __syncthreads();

  // Input-projection B rows stay in global; compiler keeps them VGPR-resident
  const uint16_t* Bx[4];
  size_t Boff[4];                                // LDS half-offsets per gate
#pragma unroll
  for (int g = 0; g < 4; ++g) {
    Bx[g]   = WxT + (size_t)(g * H_ + hc) * E_ + laneHi * 16;
    Boff[g] = (size_t)(g * 16 + lane16) * LSTRIDE + laneHi * 16;
  }
  const float bias0 = bg[hc], bias1 = bi[hc], bias2 = bf_[hc], bias3 = bo[hc];

  for (int t = 0; t < S_; ++t) {
    const uint16_t* hread  = hbf + (size_t)(t & 1) * (B_ * H_);
    uint16_t*       hwrite = hbf + (size_t)((t + 1) & 1) * (B_ * H_);

    v8f acc0 = {}, acc1 = {}, acc2 = {}, acc3 = {};

    const int tok = xTok[mrow * S_ + t];
    const uint16_t* arow = embb  + (size_t)tok  * E_ + laneHi * 16;
    const uint16_t* hrow = hread + (size_t)mrow * H_ + laneHi * 16;
    __builtin_prefetch(hrow, 0, 1);             // global_prefetch_b8 for the h row

    // A-fragment double buffer: groups of 4 K-chunks (each chunk = 32 halves)
    v16bf abuf[4], anext[4];
#pragma unroll
    for (int j = 0; j < 4; ++j) abuf[j] = load_a16(arow + j * 32);

    // ---- input projection: z += emb[x[m,t]] @ Wx   (K = 256, 2 groups) ----
#pragma unroll
    for (int j = 0; j < 4; ++j) anext[j] = load_a16(arow + 128 + j * 32);
#pragma unroll
    for (int j = 0; j < 4; ++j) {
      const int k = j * 32;
      acc0 = WMMA_BF16(abuf[j], load_b32(Bx[0] + k), acc0);
      acc1 = WMMA_BF16(abuf[j], load_b32(Bx[1] + k), acc1);
      acc2 = WMMA_BF16(abuf[j], load_b32(Bx[2] + k), acc2);
      acc3 = WMMA_BF16(abuf[j], load_b32(Bx[3] + k), acc3);
    }
#pragma unroll
    for (int j = 0; j < 4; ++j) abuf[j] = anext[j];
    // group 1 of input projection; prefetch first h group of the recurrence
#pragma unroll
    for (int j = 0; j < 4; ++j) anext[j] = load_a16(hrow + j * 32);
#pragma unroll
    for (int j = 0; j < 4; ++j) {
      const int k = 128 + j * 32;
      acc0 = WMMA_BF16(abuf[j], load_b32(Bx[0] + k), acc0);
      acc1 = WMMA_BF16(abuf[j], load_b32(Bx[1] + k), acc1);
      acc2 = WMMA_BF16(abuf[j], load_b32(Bx[2] + k), acc2);
      acc3 = WMMA_BF16(abuf[j], load_b32(Bx[3] + k), acc3);
    }
#pragma unroll
    for (int j = 0; j < 4; ++j) abuf[j] = anext[j];

    // ---- recurrence: z += h @ Wh (K = 1024, 8 groups, B from LDS) ----
#pragma unroll
    for (int g = 0; g < 8; ++g) {
      if (g < 7) {
#pragma unroll
        for (int j = 0; j < 4; ++j)
          anext[j] = load_a16(hrow + (g + 1) * 128 + j * 32);
      }
#pragma unroll
      for (int j = 0; j < 4; ++j) {
        const int k = g * 128 + j * 32;
        acc0 = WMMA_BF16(abuf[j], load_b32(&sWh[Boff[0] + k]), acc0);
        acc1 = WMMA_BF16(abuf[j], load_b32(&sWh[Boff[1] + k]), acc1);
        acc2 = WMMA_BF16(abuf[j], load_b32(&sWh[Boff[2] + k]), acc2);
        acc3 = WMMA_BF16(abuf[j], load_b32(&sWh[Boff[3] + k]), acc3);
      }
      if (g < 7) {
#pragma unroll
        for (int j = 0; j < 4; ++j) abuf[j] = anext[j];
      }
    }

    // ---- gates + state update (acc VGPR r -> M = r + 8*laneHi, N = lane%16) ----
#pragma unroll
    for (int r = 0; r < 8; ++r) {
      const int m = m0 + r + laneHi * 8;
      const float zg = acc0[r] + bias0;
      const float zi = acc1[r] + bias1;
      const float zf = acc2[r] + bias2;
      const float zo = acc3[r] + bias3;
      const float gv = tanhf(zg);
      const float iv = 1.0f / (1.0f + __expf(-zi));
      const float fv = 1.0f / (1.0f + __expf(-zf));
      const float ov = 1.0f / (1.0f + __expf(-zo));
      const size_t off = (size_t)m * H_ + hc;
      const float cn = gv * iv + cst[off] * fv;
      cst[off] = cn;
      const float hn = tanhf(cn) * ov;
      hwrite[off] = f32_to_bf16(hn);
      if (t == S_ - 1) hf32[off] = hn;
    }

    // ---- grid barrier (monotonic counter; 64 resident WGs) ----
    __threadfence();
    __syncthreads();
    if (tid == 0) {
      __hip_atomic_fetch_add(cnt, 1u, __ATOMIC_RELEASE, __HIP_MEMORY_SCOPE_AGENT);
      const unsigned tgt = (unsigned)(t + 1) * NBLK;
      while (__hip_atomic_load(cnt, __ATOMIC_ACQUIRE, __HIP_MEMORY_SCOPE_AGENT) < tgt)
        __builtin_amdgcn_s_sleep(1);
    }
    __syncthreads();
    __threadfence();
  }
}

// ---------------- head: p = h @ W_ph + b_p, log_softmax ----------------
__global__ __launch_bounds__(256)
void k_final(const float* __restrict__ hf32, const float* __restrict__ Wph,
             const float* __restrict__ bp, float* __restrict__ out) {
  __shared__ float red[256];
  __shared__ float pc[C_];
  const int b = blockIdx.x, tid = threadIdx.x;
  float p[C_];
#pragma unroll
  for (int c = 0; c < C_; ++c) p[c] = 0.0f;
  for (int k = tid; k < H_; k += 256) {
    const float hv = hf32[b * H_ + k];
#pragma unroll
    for (int c = 0; c < C_; ++c) p[c] += hv * Wph[k * C_ + c];
  }
  for (int c = 0; c < C_; ++c) {
    red[tid] = p[c]; __syncthreads();
    for (int s = 128; s > 0; s >>= 1) {
      if (tid < s) red[tid] += red[tid + s];
      __syncthreads();
    }
    if (tid == 0) pc[c] = red[0] + bp[c];
    __syncthreads();
  }
  if (tid < C_) {
    float mx = pc[0];
#pragma unroll
    for (int c = 1; c < C_; ++c) mx = fmaxf(mx, pc[c]);
    float se = 0.0f;
#pragma unroll
    for (int c = 0; c < C_; ++c) se += __expf(pc[c] - mx);
    const float lse = mx + __logf(se);
    out[b * C_ + tid] = pc[tid] - lse;
  }
}

extern "C" void kernel_launch(void* const* d_in, const int* in_sizes, int n_in,
                              void* d_out, int out_size, void* d_ws, size_t ws_size,
                              hipStream_t stream) {
  const int*   x   = (const int*)d_in[0];
  const float* emb = (const float*)d_in[1];
  const float* Wgx = (const float*)d_in[2];
  const float* Wgh = (const float*)d_in[3];
  const float* bg  = (const float*)d_in[4];
  const float* Wix = (const float*)d_in[5];
  const float* Wih = (const float*)d_in[6];
  const float* bi  = (const float*)d_in[7];
  const float* Wfx = (const float*)d_in[8];
  const float* Wfh = (const float*)d_in[9];
  const float* bf  = (const float*)d_in[10];
  const float* Wox = (const float*)d_in[11];
  const float* Woh = (const float*)d_in[12];
  const float* bo  = (const float*)d_in[13];
  const float* Wph = (const float*)d_in[14];
  const float* bp  = (const float*)d_in[15];

  char* ws = (char*)d_ws;
  size_t o = 0;
  uint16_t* WxT  = (uint16_t*)(ws + o); o += (size_t)N4_ * E_ * 2;       // 2 MB
  uint16_t* WhT  = (uint16_t*)(ws + o); o += (size_t)N4_ * H_ * 2;       // 8 MB
  uint16_t* embb = (uint16_t*)(ws + o); o += (size_t)V_ * E_ * 2;        // 64 KB
  uint16_t* hbf  = (uint16_t*)(ws + o); o += (size_t)2 * B_ * H_ * 2;    // 512 KB (double buffer)
  float*    hf32 = (float*)(ws + o);    o += (size_t)B_ * H_ * 4;        // 512 KB
  float*    cst  = (float*)(ws + o);    o += (size_t)B_ * H_ * 4;        // 512 KB
  unsigned* cnt  = (unsigned*)(ws + o); o += 256;

  k_prep_wx  <<<(N4_ * E_) / 256, 256, 0, stream>>>(Wgx, Wix, Wfx, Wox, WxT);
  k_prep_wh  <<<(N4_ * H_) / 256, 256, 0, stream>>>(Wgh, Wih, Wfh, Woh, WhT);
  k_prep_misc<<<512, 256, 0, stream>>>(emb, embb, hbf, hf32, cst, cnt);
  k_lstm     <<<NBLK, 256, 0, stream>>>(x, embb, WxT, WhT, bg, bi, bf, bo,
                                        hbf, hf32, cst, cnt);
  k_final    <<<B_, 256, 0, stream>>>(hf32, Wph, bp, (float*)d_out);
}
